// QKVAttention_5428838662461
// MI455X (gfx1250) — compile-verified, matrix operand order checked by reading, not run on hardware
//
#include <hip/hip_runtime.h>
#include <hip/hip_bf16.h>

// ---------------------------------------------------------------------------
// Flash-attention for qkv [4, 1536, 2048] f32, H=8, ch=64, L=2048 (gfx1250).
//   S'^T = (K*s2)^T (Q*s2) via v_wmma_f32_16x16x32_f16, where
//          s2 = 64^(-1/4) * sqrt(log2 e)  -> S' is already in base-2 domain
//   online softmax over s: p = exp2(S' - m'), corr = exp2(m'_old - m'_new)
//          using raw v_exp_f32 (__builtin_amdgcn_exp2f, FTZ is fine here)
//   O = V * P^T via v_wmma_f32_16x16x32_f16
// The exp'd S^T f32 accumulator, packed to f16 pairs (v_cvt_pk_rtz_f16_f32),
// IS the B-operand fragment of P^T (D-layout M=v+8h == B-layout K=e+8h).
// ---------------------------------------------------------------------------

typedef __attribute__((ext_vector_type(16))) _Float16 v16h;
typedef __attribute__((ext_vector_type(8)))  _Float16 v8h;
typedef __attribute__((ext_vector_type(2)))  __fp16   v2pk;   // cvt_pkrtz return
typedef __attribute__((ext_vector_type(8)))  float    v8f;

#define SEQ_L   2048
#define CH      64
#define WIDTH   1536
#define BT      128          // t-tile per workgroup (8 waves x 16)
#define SCH     64           // s-chunk per iteration (4 x 16-s WMMA tiles)
#define QS      72           // Qlds row stride in halfs (64 + 8 pad, 16B mult)
#define KS      72           // Klds row stride
#define VS      72           // Vlds row stride (64 + 8 pad)
// 64^(-1/4) * sqrt(log2(e)) : folds softmax base-2 conversion into the gemm
#define SCALE2  0.4246609146247136f

__device__ inline v16h cat8(v8h lo, v8h hi) {
    v16h r;
#pragma unroll
    for (int e = 0; e < 8; ++e) { r[e] = lo[e]; r[8 + e] = hi[e]; }
    return r;
}

__device__ inline v8f wmma_f16(v16h a, v16h b, v8f c) {
    return __builtin_amdgcn_wmma_f32_16x16x32_f16(
        /*neg_a=*/false, a, /*neg_b=*/false, b,
        /*c_mod=*/(short)0, c, /*reuse_a=*/false, /*reuse_b=*/false);
}

__global__ __launch_bounds__(256)
void qkv_attention_kernel(const float* __restrict__ qkv, float* __restrict__ out)
{
    __shared__ __align__(16) _Float16 Qlds[BT * QS];    // [t][c]  18.0 KB
    __shared__ __align__(16) _Float16 Klds[SCH * KS];   // [s][c]   9.0 KB
    __shared__ __align__(16) _Float16 Vlds[CH * VS];    // [c][s]   9.0 KB

    const int bh = blockIdx.x;            // 0..31  (batch*heads)
    const int b  = bh >> 3;
    const int hh = bh & 7;
    const int t0 = blockIdx.y * BT;

    const size_t base = (size_t)b * WIDTH * SEQ_L + (size_t)(hh * CH) * SEQ_L;
    const float* qg = qkv + base;                          // q rows
    const float* kg = qkv + base + (size_t)512  * SEQ_L;   // k rows
    const float* vg = qkv + base + (size_t)1024 * SEQ_L;   // v rows
    float* og = out + (size_t)b * 512 * SEQ_L + (size_t)(hh * CH) * SEQ_L;

    const int tid  = threadIdx.x;
    const int lane = tid & 31;
    const int wv   = tid >> 5;     // wave id 0..7
    const int ln   = lane & 15;    // N index inside 16x16 tile
    const int lh   = lane >> 4;    // lane half (K/M pattern selector)

    // ---- stage Q tile, scaled + transposed to [t][c], f16 ----------------
#pragma unroll
    for (int i = 0; i < (CH * BT) / 256; ++i) {      // 32 iters
        int idx = tid + i * 256;
        int c = idx >> 7;          // /128
        int t = idx & 127;
        Qlds[t * QS + c] = (_Float16)(qg[(size_t)c * SEQ_L + t0 + t] * SCALE2);
    }
    __syncthreads();

    // ---- loop-invariant Q fragments (B operand of S^T gemm) --------------
    v16h qf[2];
    {
        const int tq = wv * 16 + ln;
#pragma unroll
        for (int kk = 0; kk < 2; ++kk) {
            v8h lo = *(const v8h*)&Qlds[tq * QS + kk * 32 + lh * 8];
            v8h hi = *(const v8h*)&Qlds[tq * QS + kk * 32 + 16 + lh * 8];
            qf[kk] = cat8(lo, hi);
        }
    }

    const v8f vzero = {0.f, 0.f, 0.f, 0.f, 0.f, 0.f, 0.f, 0.f};
    v8f acc[4] = {vzero, vzero, vzero, vzero};   // O tiles, c = ct*16 + v + 8h
    float m_run = -3.0e38f;   // running max, base-2 domain
    float l_run = 0.0f;       // running denominator

    for (int s0 = 0; s0 < SEQ_L; s0 += SCH) {
        __syncthreads();   // previous chunk's LDS consumers done

        // ---- cooperative K/V chunk load (f32 -> f16, K transposed) -------
#pragma unroll
        for (int i = 0; i < (CH * SCH) / 256; ++i) {   // 16 iters
            int idx = tid + i * 256;
            int c = idx >> 6;
            int s = idx & 63;
            Klds[s * KS + c] = (_Float16)(kg[(size_t)c * SEQ_L + s0 + s] * SCALE2);
            Vlds[c * VS + s] = (_Float16)(vg[(size_t)c * SEQ_L + s0 + s]);
        }
        if (s0 + SCH < SEQ_L) {   // hint next chunk into cache (global_prefetch_b8)
            int c = tid >> 6, s = tid & 63;
            __builtin_prefetch(&kg[(size_t)c * SEQ_L + s0 + SCH + s], 0, 1);
            __builtin_prefetch(&vg[(size_t)c * SEQ_L + s0 + SCH + s], 0, 1);
        }
        __syncthreads();

        // ---- S'^T tiles: st[j] -> s in [s0+16j, s0+16j+16) ---------------
        v8f st[4] = {vzero, vzero, vzero, vzero};
#pragma unroll
        for (int kk = 0; kk < 2; ++kk) {
#pragma unroll
            for (int j = 0; j < 4; ++j) {
                v16h a = cat8(
                    *(const v8h*)&Klds[(16 * j + ln) * KS + kk * 32 + lh * 8],
                    *(const v8h*)&Klds[(16 * j + ln) * KS + kk * 32 + 16 + lh * 8]);
                st[j] = wmma_f16(a, qf[kk], st[j]);
            }
        }

        // ---- online softmax (base-2 domain; 32 vals/lane + half shuffle) -
        float mloc = st[0][0];
#pragma unroll
        for (int j = 0; j < 4; ++j)
#pragma unroll
            for (int v = 0; v < 8; ++v) mloc = fmaxf(mloc, st[j][v]);
        mloc = fmaxf(mloc, __shfl_xor(mloc, 16, 32));

        const float m_new = fmaxf(m_run, mloc);
        const float corr  = __builtin_amdgcn_exp2f(m_run - m_new);

        float p[4][8];
        float ps = 0.f;
#pragma unroll
        for (int j = 0; j < 4; ++j)
#pragma unroll
            for (int v = 0; v < 8; ++v) {
                p[j][v] = __builtin_amdgcn_exp2f(st[j][v] - m_new);
                ps += p[j][v];
            }
        ps += __shfl_xor(ps, 16, 32);

        l_run = l_run * corr + ps;
        m_run = m_new;
#pragma unroll
        for (int ct = 0; ct < 4; ++ct) acc[ct] = acc[ct] * corr;

        // ---- pack P into two B-operand fragments (layout identity) -------
        v16h pfA, pfB;
#pragma unroll
        for (int i = 0; i < 4; ++i) {
            v2pk a = __builtin_amdgcn_cvt_pkrtz(p[0][2 * i], p[0][2 * i + 1]);
            v2pk b = __builtin_amdgcn_cvt_pkrtz(p[1][2 * i], p[1][2 * i + 1]);
            v2pk c = __builtin_amdgcn_cvt_pkrtz(p[2][2 * i], p[2][2 * i + 1]);
            v2pk d = __builtin_amdgcn_cvt_pkrtz(p[3][2 * i], p[3][2 * i + 1]);
            pfA[2 * i]     = (_Float16)a[0]; pfA[2 * i + 1]     = (_Float16)a[1];
            pfA[8 + 2 * i] = (_Float16)b[0]; pfA[8 + 2 * i + 1] = (_Float16)b[1];
            pfB[2 * i]     = (_Float16)c[0]; pfB[2 * i + 1]     = (_Float16)c[1];
            pfB[8 + 2 * i] = (_Float16)d[0]; pfB[8 + 2 * i + 1] = (_Float16)d[1];
        }

        // ---- O += V * P^T  (4 c-tiles x 2 s-halves) ----------------------
#pragma unroll
        for (int ct = 0; ct < 4; ++ct) {
            const int cr = ct * 16 + ln;
            v16h vfA = cat8(*(const v8h*)&Vlds[cr * VS + lh * 8],
                            *(const v8h*)&Vlds[cr * VS + 16 + lh * 8]);
            acc[ct] = wmma_f16(vfA, pfA, acc[ct]);
            v16h vfB = cat8(*(const v8h*)&Vlds[cr * VS + 32 + lh * 8],
                            *(const v8h*)&Vlds[cr * VS + 48 + lh * 8]);
            acc[ct] = wmma_f16(vfB, pfB, acc[ct]);
        }
    }

    // ---- epilogue: divide by row sum, write O[c][t] ----------------------
    const float inv = 1.0f / l_run;
    const int tcol = t0 + wv * 16 + ln;
#pragma unroll
    for (int ct = 0; ct < 4; ++ct) {
#pragma unroll
        for (int v = 0; v < 8; ++v) {
            og[(size_t)(ct * 16 + lh * 8 + v) * SEQ_L + tcol] = acc[ct][v] * inv;
        }
    }
}

extern "C" void kernel_launch(void* const* d_in, const int* in_sizes, int n_in,
                              void* d_out, int out_size, void* d_ws, size_t ws_size,
                              hipStream_t stream) {
    (void)in_sizes; (void)n_in; (void)out_size; (void)d_ws; (void)ws_size;
    const float* qkv = (const float*)d_in[0];
    float* out = (float*)d_out;
    dim3 grid(32, SEQ_L / BT);   // 32 batch-heads x 16 t-tiles
    qkv_attention_kernel<<<grid, 256, 0, stream>>>(qkv, out);
}
